// LNN_27084063768951
// MI455X (gfx1250) — compile-verified, matrix operand order checked by reading, not run on hardware
//
#include <hip/hip_runtime.h>

#define DEVINL __device__ __forceinline__

typedef __attribute__((ext_vector_type(16))) __bf16 v16bf;
typedef __attribute__((ext_vector_type(8)))  __bf16 v8bf;
typedef __attribute__((ext_vector_type(8)))  float  v8f;
typedef __attribute__((ext_vector_type(4)))  unsigned int v4u;
typedef __attribute__((ext_vector_type(8)))  int v8i;
typedef __attribute__((ext_vector_type(4)))  int v4i;

#ifndef __has_builtin
#define __has_builtin(x) 0
#endif
#if __has_builtin(__builtin_amdgcn_tensor_load_to_lds) && \
    __has_builtin(__builtin_amdgcn_s_wait_tensorcnt)
#define LNN_TDM 1
#else
#define LNN_TDM 0
#endif

// Problem constants: B=8192 samples, hidden H=1024, z-dim = 16 (q:8, qd:8)
#define LNN_B  8192
#define LNN_H  1024
#define LNN_Z  16

DEVINL int kslot(int v, int hf) {
  // CDNA5 16-bit A/B K-slot map: VGPR v holds K pair (base, base+1)
  return ((v & 3) << 1) + ((v >> 2) << 4) + (hf << 3);
}

DEVINL float sigmoidf_(float a) { return 1.0f / (1.0f + __expf(-a)); }
DEVINL float softplusf_(float a) {
  float l = log1pf(__expf(-fabsf(a)));
  return a > 0.0f ? a + l : l;
}

DEVINL v8f wmma_bf16(v16bf a, v16bf b, v8f c) {
  return __builtin_amdgcn_wmma_f32_16x16x32_bf16(false, a, false, b, (short)0, c,
                                                 false, false);
}

// A-fragment (16x32 window, row-major, 16-bit): lane r=lane&15 holds row r.
// Each lane's 16 elements are two contiguous 8-elem runs at K=hf*8 and 16+hf*8.
DEVINL v16bf frag_rowmajor(const __bf16* p, int stride, int lane) {
  int r = lane & 15, hf = lane >> 4;
  const __bf16* rp = p + (size_t)r * stride + hf * 8;
  v8bf lo = *(const v8bf*)rp;
  v8bf hi = *(const v8bf*)(rp + 16);
  return __builtin_shufflevector(lo, hi, 0, 1, 2, 3, 4, 5, 6, 7,
                                 8, 9, 10, 11, 12, 13, 14, 15);
}

// Same as frag_rowmajor but with an elementwise bf16 scale vector (same K window).
DEVINL v16bf frag_rowmajor_scaled(const __bf16* p, int stride,
                                  const __bf16* scale, int lane) {
  int r = lane & 15, hf = lane >> 4;
  const __bf16* rp = p + (size_t)r * stride + hf * 8;
  const __bf16* sp = scale + hf * 8;
  v8bf alo = *(const v8bf*)rp,        ahi = *(const v8bf*)(rp + 16);
  v8bf slo = *(const v8bf*)sp,        shi = *(const v8bf*)(sp + 16);
  v16bf f;
#pragma unroll
  for (int e = 0; e < 8; ++e) {
    f[e]     = (__bf16)((float)alo[e] * (float)slo[e]);
    f[8 + e] = (__bf16)((float)ahi[e] * (float)shi[e]);
  }
  return f;
}

// B-fragment from a pre-swizzled buffer: one aligned 32-byte load per lane.
DEVINL v16bf frag_packed(const __bf16* pbase, int tile, int lane) {
  return *(const v16bf*)(pbase + (size_t)tile * 512 + lane * 16);
}

// One K-step's worth of fragments: A (16x32) + four packed B tiles.
struct Frags5 {
  v16bf a, b0, b1, b2, b3;
};
DEVINL Frags5 load5(const __bf16* ap, const __bf16* bp, int lane) {
  Frags5 f;
  f.a  = frag_rowmajor(ap, LNN_H, lane);
  f.b0 = frag_packed(bp, 0, lane);
  f.b1 = frag_packed(bp, 1, lane);
  f.b2 = frag_packed(bp, 2, lane);
  f.b3 = frag_packed(bp, 3, lane);
  return f;
}

#if LNN_TDM
// Generic LDS pointers carry the LDS byte offset in their low 32 bits.
DEVINL unsigned ldsoff_(const void* p) { return (unsigned)(uintptr_t)p; }

// TDM: 1-D tile DMA global -> LDS. nElems8 elements of 8 bytes, contiguous.
// D# per CDNA5 ISA: group0 = {count=1, lds_addr,
// global_addr[56:0], type=2}; group1 = {data_size=8B, tensor_dim0, tensor_dim1=1,
// tile_dim0, dim0_stride}; groups 2/3 zero (<=2D tensor).
DEVINL void tdm_load_1d(unsigned lds_byte_off, const void* gsrc, int nElems8) {
  unsigned long long ga = (unsigned long long)(uintptr_t)gsrc;
  v4u g0 = {1u,                                   // count=1 (valid descriptor)
            lds_byte_off,                         // lds_addr
            (unsigned)(ga & 0xffffffffu),         // global_addr lo
            (unsigned)(((ga >> 32) & 0x1ffffffu) | 0x80000000u)}; // hi | type=2
  v8i g1 = {(int)(3u << 16),                       // data_size = 8 bytes
            (int)(((unsigned)nElems8 & 0xffffu) << 16),   // tensor_dim0 lo
            (int)((((unsigned)nElems8 >> 16) & 0xffffu) | (1u << 16)), // | dim1=1
            (int)(((unsigned)nElems8 & 0xffffu) << 16),   // tile_dim0
            0,                                     // tile_dim1=0, tile_dim2=0
            nElems8,                               // tensor_dim0_stride
            0, 0};
  v4i gz = {0, 0, 0, 0};
#if __clang_major__ >= 23
  v8i gz8 = {0, 0, 0, 0, 0, 0, 0, 0};
  __builtin_amdgcn_tensor_load_to_lds(g0, g1, gz, gz, gz8, 0);
#else
  __builtin_amdgcn_tensor_load_to_lds(g0, g1, gz, gz, 0);
#endif
}
#endif  // LNN_TDM

// ---------------------------------------------------------------- kernel 0
// Pre-swizzle W2 (and W2^T) into WMMA B-fragment order; also W1 -> bf16.
__global__ void lnn_convert(const float* __restrict__ W2,
                            const float* __restrict__ W1,
                            __bf16* __restrict__ W2p,
                            __bf16* __restrict__ W2Tp,
                            __bf16* __restrict__ W1bf) {
  int t = blockIdx.x * blockDim.x + threadIdx.x;   // 0 .. 1M-1
  int e    = t & 15;
  int lane = (t >> 4) & 31;
  int ct   = (t >> 9) & 63;
  int kt   = t >> 15;
  int kk = kt * 32 + kslot(e >> 1, lane >> 4) + (e & 1);
  int nn = ct * 16 + (lane & 15);
  W2p[t]  = (__bf16)W2[(size_t)kk * LNN_H + nn];
  W2Tp[t] = (__bf16)W2[(size_t)nn * LNN_H + kk];
  if (t < LNN_Z * LNN_H) W1bf[t] = (__bf16)W1[t];
}

// ---------------------------------------------------------------- kernel 1
// a1 = z @ W1 + b1 ; h1 = softplus(a1)
__global__ void lnn_fwd1(const float* __restrict__ x, const float* __restrict__ W1,
                         const float* __restrict__ b1, float* __restrict__ a1,
                         __bf16* __restrict__ h1bf) {
  int t = blockIdx.x * blockDim.x + threadIdx.x;   // 0 .. B*H-1
  int b = t >> 10, i = t & 1023;
  const float* xr = x + b * LNN_Z;
  float a = b1[i];
#pragma unroll
  for (int k = 0; k < LNN_Z; ++k) a += xr[k] * W1[k * LNN_H + i];
  a1[(size_t)t]   = a;
  h1bf[(size_t)t] = (__bf16)softplusf_(a);
}

// ---------------------------------------------------------------- kernel 2
// a2 = h1 @ W2 + b2 ; fused epilogue: u = W3*sigma(a2), d2 = W3*sigma'(a2)
__global__ void lnn_gemm_a2(const __bf16* __restrict__ h1bf,
                            const __bf16* __restrict__ W2p,
                            const float* __restrict__ b2,
                            const float* __restrict__ W3,
                            __bf16* __restrict__ ubf, __bf16* __restrict__ d2bf) {
  int lane = threadIdx.x & 31;
  int job  = blockIdx.x * 8 + (threadIdx.x >> 5);  // 8192 wave-jobs
  int r0   = (job & 511) * 16;                     // sample strip
  int ct0  = (job >> 9) * 4;                       // first 16-col tile
  v8f zero = {0.f, 0.f, 0.f, 0.f, 0.f, 0.f, 0.f, 0.f};
  v8f z0 = zero, z1 = zero, z2 = zero, z3 = zero;

  const __bf16* ap = h1bf + (size_t)r0 * LNN_H;
  const __bf16* bp = W2p + (size_t)ct0 * 512;
  Frags5 cur = load5(ap, bp, lane);
#pragma unroll
  for (int kt = 0; kt < 31; ++kt) {
    ap += 32;
    bp += (size_t)64 * 512;
    Frags5 nxt = load5(ap, bp, lane);
    z0 = wmma_bf16(cur.a, cur.b0, z0);
    z1 = wmma_bf16(cur.a, cur.b1, z1);
    z2 = wmma_bf16(cur.a, cur.b2, z2);
    z3 = wmma_bf16(cur.a, cur.b3, z3);
    cur = nxt;
  }
  z0 = wmma_bf16(cur.a, cur.b0, z0);
  z1 = wmma_bf16(cur.a, cur.b1, z1);
  z2 = wmma_bf16(cur.a, cur.b2, z2);
  z3 = wmma_bf16(cur.a, cur.b3, z3);

  int n = lane & 15, hf = lane >> 4;
  v8f zz[4] = {z0, z1, z2, z3};
#pragma unroll
  for (int t = 0; t < 4; ++t) {
    int col = (ct0 + t) * 16 + n;
    float w3 = W3[col], bb = b2[col];
#pragma unroll
    for (int i = 0; i < 8; ++i) {
      int row  = r0 + i + 8 * hf;
      float a2 = zz[t][i] + bb;
      float s  = sigmoidf_(a2);
      ubf[(size_t)row * LNN_H + col]  = (__bf16)(w3 * s);
      d2bf[(size_t)row * LNN_H + col] = (__bf16)(w3 * s * (1.f - s));
    }
  }
}

// ---------------------------------------------------------------- kernel 3
// m = u @ W2^T ; fused epilogue: d1 = sigma'(a1)*m, v = sigma(a1)*m
__global__ void lnn_gemm_w2u(const __bf16* __restrict__ ubf,
                             const __bf16* __restrict__ W2Tp,
                             const float* __restrict__ a1,
                             __bf16* __restrict__ d1bf, __bf16* __restrict__ vbf) {
  int lane = threadIdx.x & 31;
  int job  = blockIdx.x * 8 + (threadIdx.x >> 5);
  int r0   = (job & 511) * 16;
  int ct0  = (job >> 9) * 4;
  v8f zero = {0.f, 0.f, 0.f, 0.f, 0.f, 0.f, 0.f, 0.f};
  v8f z0 = zero, z1 = zero, z2 = zero, z3 = zero;

  const __bf16* ap = ubf + (size_t)r0 * LNN_H;
  const __bf16* bp = W2Tp + (size_t)ct0 * 512;
  Frags5 cur = load5(ap, bp, lane);
#pragma unroll
  for (int kt = 0; kt < 31; ++kt) {
    ap += 32;
    bp += (size_t)64 * 512;
    Frags5 nxt = load5(ap, bp, lane);
    z0 = wmma_bf16(cur.a, cur.b0, z0);
    z1 = wmma_bf16(cur.a, cur.b1, z1);
    z2 = wmma_bf16(cur.a, cur.b2, z2);
    z3 = wmma_bf16(cur.a, cur.b3, z3);
    cur = nxt;
  }
  z0 = wmma_bf16(cur.a, cur.b0, z0);
  z1 = wmma_bf16(cur.a, cur.b1, z1);
  z2 = wmma_bf16(cur.a, cur.b2, z2);
  z3 = wmma_bf16(cur.a, cur.b3, z3);

  int n = lane & 15, hf = lane >> 4;
  v8f zz[4] = {z0, z1, z2, z3};
#pragma unroll
  for (int t = 0; t < 4; ++t) {
    int col = (ct0 + t) * 16 + n;
#pragma unroll
    for (int i = 0; i < 8; ++i) {
      int row  = r0 + i + 8 * hf;
      float m  = zz[t][i];
      float s  = sigmoidf_(a1[(size_t)row * LNN_H + col]);
      d1bf[(size_t)row * LNN_H + col] = (__bf16)(s * (1.f - s) * m);
      vbf[(size_t)row * LNN_H + col]  = (__bf16)(s * m);
    }
  }
}

// ---------------------------------------------------------------- kernel 4
// g[b][k] = sum_i W1[k][i] * v[b][i]   (one wave32 per sample, shuffle reduce)
__global__ void lnn_grad_z(const __bf16* __restrict__ vbf,
                           const float* __restrict__ W1, float* __restrict__ g) {
  int lane = threadIdx.x & 31;
  int b    = blockIdx.x * 8 + (threadIdx.x >> 5);
  float acc[LNN_Z];
#pragma unroll
  for (int k = 0; k < LNN_Z; ++k) acc[k] = 0.f;
  for (int i = lane; i < LNN_H; i += 32) {
    float vv = (float)vbf[(size_t)b * LNN_H + i];
#pragma unroll
    for (int k = 0; k < LNN_Z; ++k) acc[k] += W1[k * LNN_H + i] * vv;
  }
#pragma unroll
  for (int k = 0; k < LNN_Z; ++k) {
    acc[k] += __shfl_xor(acc[k], 16, 32);
    acc[k] += __shfl_xor(acc[k], 8, 32);
    acc[k] += __shfl_xor(acc[k], 4, 32);
    acc[k] += __shfl_xor(acc[k], 2, 32);
    acc[k] += __shfl_xor(acc[k], 1, 32);
  }
  if (lane == 0) {
#pragma unroll
    for (int k = 0; k < LNN_Z; ++k) g[b * LNN_Z + k] = acc[k];
  }
}

// ---------------------------------------------------------------- kernel 5
// Per-sample: H = W1 diag(d1) W1^T + Q diag(d2) Q^T with Q = (W1*sigma(a1)) @ W2.
// 4 waves/sample. B tiles of the Q GEMM are double-buffered into LDS by the
// Tensor Data Mover (TENSORcnt), decoupling the W2 stream from the WMMA pipe.
__global__ __launch_bounds__(128) void lnn_hessian_solve(
    const float* __restrict__ x, const float* __restrict__ W1,
    const __bf16* __restrict__ W1bf, const float* __restrict__ a1,
    const __bf16* __restrict__ W2p, const __bf16* __restrict__ d1bf,
    const __bf16* __restrict__ d2bf, const float* __restrict__ g,
    float* __restrict__ out) {
  __shared__ __align__(16) __bf16 Pl[LNN_Z * LNN_H];   // 32 KB: P = W1*sigma(a1)
  __shared__ __align__(16) __bf16 d1l[LNN_H];          // 2 KB
  __shared__ __align__(16) __bf16 d2l[LNN_H];          // 2 KB
  __shared__ __align__(16) __bf16 sb[4][2][1024];      // 16 KB: per-wave stage bufs
  __shared__ float Hl[LNN_Z * LNN_Z];                  // 1 KB

  int tid = threadIdx.x;
  int b   = blockIdx.x;
  for (int i = tid; i < LNN_H; i += 128) {
    d1l[i] = d1bf[(size_t)b * LNN_H + i];
    d2l[i] = d2bf[(size_t)b * LNN_H + i];
  }
  for (int t = tid; t < LNN_Z * LNN_H; t += 128) {
    int i = t & 1023;
    Pl[t] = (__bf16)(W1[t] * sigmoidf_(a1[(size_t)b * LNN_H + i]));
  }
  for (int i = tid; i < LNN_Z * LNN_Z; i += 128) Hl[i] = 0.f;
  __syncthreads();

  int lane = tid & 31, wave = tid >> 5;
  int hf = lane >> 4;
  v8f zero = {0.f, 0.f, 0.f, 0.f, 0.f, 0.f, 0.f, 0.f};
  v8f hacc = zero;

  // --- term 2: each wave owns 8 chunks of 32 Q-columns (2 packed B tiles).
  for (int c = 0; c < 8; ++c) {
    int c0  = wave * 256 + c * 32;
    int ct0 = c0 >> 4;
    v8f z0 = zero, z1 = zero;
    __bf16* sb0 = &sb[wave][0][0];
    __bf16* sb1 = &sb[wave][1][0];
#if LNN_TDM
    tdm_load_1d(ldsoff_(sb0), W2p + (size_t)ct0 * 512, 256);  // kt=0, 2 KB
#endif
    for (int kt = 0; kt < 32; ++kt) {
#if LNN_TDM
      __bf16* cb = (kt & 1) ? sb1 : sb0;
      __bf16* nb = (kt & 1) ? sb0 : sb1;
      if (kt < 31) {
        tdm_load_1d(ldsoff_(nb), W2p + ((size_t)(kt + 1) * 64 + ct0) * 512, 256);
        __builtin_amdgcn_s_wait_tensorcnt(1);  // kt's DMA done, kt+1 in flight
      } else {
        __builtin_amdgcn_s_wait_tensorcnt(0);
      }
      v16bf b0 = frag_packed(cb, 0, lane);
      v16bf b1 = frag_packed(cb, 1, lane);
#else
      const __bf16* gb = W2p + ((size_t)kt * 64 + ct0) * 512;
      v16bf b0 = frag_packed(gb, 0, lane);
      v16bf b1 = frag_packed(gb, 1, lane);
#endif
      v16bf af = frag_rowmajor(Pl + kt * 32, LNN_H, lane);   // A = P (LDS)
      z0 = wmma_bf16(af, b0, z0);
      z1 = wmma_bf16(af, b1, z1);
    }
#if LNN_TDM
    __builtin_amdgcn_s_wait_tensorcnt(0);  // before reusing stage bufs below
#endif
    // Spill Q (and Q*d2) through LDS (reusing the stage buffers): in-wave lane
    // transpose for the rank-32 update  H += (Q*d2) @ Q^T  (one K=32 WMMA).
    {
      int n = lane & 15;
      float d0  = (float)d2l[c0 + n];
      float d1v = (float)d2l[c0 + 16 + n];
#pragma unroll
      for (int i = 0; i < 8; ++i) {
        int row = i + 8 * hf;
        sb0[row * 32 + n]      = (__bf16)z0[i];            // ql
        sb0[row * 32 + 16 + n] = (__bf16)z1[i];
        sb1[row * 32 + n]      = (__bf16)(z0[i] * d0);     // qdl
        sb1[row * 32 + 16 + n] = (__bf16)(z1[i] * d1v);
      }
    }
    v16bf aq = frag_rowmajor(sb1, 32, lane);  // A = Q*d2 (16x32)
    v16bf bq = frag_rowmajor(sb0, 32, lane);  // B = Q^T (row-major read)
    hacc = wmma_bf16(aq, bq, hacc);
  }

  // --- term 1: H += W1 diag(d1) W1^T ; each wave owns 8 K-tiles of 32.
#pragma unroll 4
  for (int kt = wave * 8; kt < wave * 8 + 8; ++kt) {
    v16bf af = frag_rowmajor_scaled(W1bf + kt * 32, LNN_H, d1l + kt * 32, lane);
    v16bf bw = frag_rowmajor(W1bf + kt * 32, LNN_H, lane);  // B[i][n] = W1[n][i]
    hacc = wmma_bf16(af, bw, hacc);
  }

  // Combine partial H across the 4 waves (ds_add_f32 atomics on LDS).
  {
    int n = lane & 15;
#pragma unroll
    for (int i = 0; i < 8; ++i) {
      int row = i + 8 * hf;
      atomicAdd(&Hl[row * LNN_Z + n], hacc[i]);
    }
  }
  __syncthreads();

  // --- 8x8 pivoted Gauss-Jordan solve:  M qdd = g_q - C qd
  if (tid == 0) {
    float Mm[8][8], rhs[8], qd[8];
#pragma unroll
    for (int j = 0; j < 8; ++j) qd[j] = x[b * LNN_Z + 8 + j];
    for (int i = 0; i < 8; ++i) {
      float s = g[b * LNN_Z + i];
      for (int j = 0; j < 8; ++j) s -= Hl[(8 + i) * LNN_Z + j] * qd[j];
      rhs[i] = s;
      for (int j = 0; j < 8; ++j) Mm[i][j] = Hl[(8 + i) * LNN_Z + 8 + j];
    }
    for (int c = 0; c < 8; ++c) {
      int p = c;
      float best = fabsf(Mm[c][c]);
      for (int rr = c + 1; rr < 8; ++rr) {
        float vv = fabsf(Mm[rr][c]);
        if (vv > best) { best = vv; p = rr; }
      }
      if (p != c) {
        for (int j = 0; j < 8; ++j) { float t = Mm[c][j]; Mm[c][j] = Mm[p][j]; Mm[p][j] = t; }
        float t = rhs[c]; rhs[c] = rhs[p]; rhs[p] = t;
      }
      float inv = 1.0f / Mm[c][c];
      for (int j = c; j < 8; ++j) Mm[c][j] *= inv;
      rhs[c] *= inv;
      for (int rr = 0; rr < 8; ++rr) {
        if (rr == c) continue;
        float f = Mm[rr][c];
        for (int j = c; j < 8; ++j) Mm[rr][j] -= f * Mm[c][j];
        rhs[rr] -= f * rhs[c];
      }
    }
    for (int k = 0; k < 8; ++k) {
      out[b * LNN_Z + k]     = qd[k];
      out[b * LNN_Z + 8 + k] = rhs[k];
    }
  }
}

extern "C" void kernel_launch(void* const* d_in, const int* in_sizes, int n_in,
                              void* d_out, int out_size, void* d_ws, size_t ws_size,
                              hipStream_t stream) {
  (void)in_sizes; (void)n_in; (void)out_size; (void)ws_size;
  const float* x  = (const float*)d_in[0];
  const float* W1 = (const float*)d_in[1];
  const float* b1 = (const float*)d_in[2];
  const float* W2 = (const float*)d_in[3];
  const float* b2 = (const float*)d_in[4];
  const float* W3 = (const float*)d_in[5];
  // b3 (d_in[6]) does not affect any derivative.
  float* out = (float*)d_out;

  char* ws = (char*)d_ws;
  size_t o = 0;
  __bf16* W2p   = (__bf16*)(ws + o); o += (size_t)LNN_H * LNN_H * 2;
  __bf16* W2Tp  = (__bf16*)(ws + o); o += (size_t)LNN_H * LNN_H * 2;
  __bf16* W1bf  = (__bf16*)(ws + o); o += (size_t)LNN_Z * LNN_H * 2;
  float*  a1    = (float*) (ws + o); o += (size_t)LNN_B * LNN_H * 4;
  __bf16* h1bf  = (__bf16*)(ws + o); o += (size_t)LNN_B * LNN_H * 2;
  __bf16* ubf   = (__bf16*)(ws + o); o += (size_t)LNN_B * LNN_H * 2;
  __bf16* d2bf  = (__bf16*)(ws + o); o += (size_t)LNN_B * LNN_H * 2;
  __bf16* d1bf  = (__bf16*)(ws + o); o += (size_t)LNN_B * LNN_H * 2;
  __bf16* vbf   = (__bf16*)(ws + o); o += (size_t)LNN_B * LNN_H * 2;
  float*  g     = (float*) (ws + o); o += (size_t)LNN_B * LNN_Z * 4;

  lnn_convert      <<<(LNN_H * LNN_H) / 256, 256, 0, stream>>>(W2, W1, W2p, W2Tp, W1bf);
  lnn_fwd1         <<<(LNN_B * LNN_H) / 256, 256, 0, stream>>>(x, W1, b1, a1, h1bf);
  lnn_gemm_a2      <<<1024, 256, 0, stream>>>(h1bf, W2p, b2, W3, ubf, d2bf);
  lnn_gemm_w2u     <<<1024, 256, 0, stream>>>(ubf, W2Tp, a1, d1bf, vbf);
  lnn_grad_z       <<<1024, 256, 0, stream>>>(vbf, W1, g);
  lnn_hessian_solve<<<LNN_B, 128, 0, stream>>>(x, W1, W1bf, a1, W2p, d1bf, d2bf, g, out);
}
